// SEIRPhysicsPINN_88252987998857
// MI455X (gfx1250) — matrix operand order repaired and checked by reading, not compiled, not run on hardware
//
#include <hip/hip_runtime.h>
#include <hip/hip_bf16.h>
#include <math.h>

// Problem constants (from reference setup_inputs)
#define BB   64
#define TT   256
#define NN   2048
#define HOR  16
#define NSTEPS (TT + HOR - 1)          // 271
#define BN   (BB * NN)                 // 131072
#define OUT_I_ELEMS ((size_t)BB * HOR * NN)   // 2,097,152 floats

// Workspace layout (bytes)
#define OFF_SYNC  ((size_t)0)                       // 256 B reserved
#define OFF_S     ((size_t)256)
#define OFF_E     (OFF_S   + (size_t)BN * 4)
#define OFF_I     (OFF_E   + (size_t)BN * 4)
#define OFF_IBF0  (OFF_I   + (size_t)BN * 4)
#define OFF_IBF1  (OFF_IBF0 + (size_t)BN * 2)
#define OFF_PIT   (OFF_IBF1 + (size_t)BN * 2)       // 2048*2048 bf16 = 8 MB

#define NBLK 64
#define LDSB_STRIDE 2056   // 2048 + 8 bf16 pad -> row stride 4112 B = 1028 dw == 4 mod 64 banks

typedef __attribute__((ext_vector_type(16))) __bf16 v16bf;
typedef __attribute__((ext_vector_type(8)))  __bf16 v8bf;
typedef __attribute__((ext_vector_type(8)))  float  v8f;

// ---------------------------------------------------------------------------
// Kernel 0: zero the grid-sync counters (workspace is poisoned by harness)
// ---------------------------------------------------------------------------
__global__ void seir_zero_sync(unsigned* p) {
    if (threadIdx.x < 8) p[threadIdx.x] = 0u;
}

// ---------------------------------------------------------------------------
// Kernel 1: row softmax of pi_logits -> f32 Pi written directly into d_out
// ---------------------------------------------------------------------------
__global__ __launch_bounds__(256) void seir_softmax(const float* __restrict__ logits,
                                                    float* __restrict__ Pi) {
    __shared__ float red[256];
    const int row = blockIdx.x;
    const int tid = threadIdx.x;
    const float* src = logits + (size_t)row * NN;

    float v[8];
    float vmax = -INFINITY;
#pragma unroll
    for (int j = 0; j < 8; ++j) {
        v[j] = src[tid + j * 256];
        vmax = fmaxf(vmax, v[j]);
    }
    red[tid] = vmax;
    __syncthreads();
    for (int s = 128; s > 0; s >>= 1) {
        if (tid < s) red[tid] = fmaxf(red[tid], red[tid + s]);
        __syncthreads();
    }
    vmax = red[0];
    __syncthreads();

    float sum = 0.f;
#pragma unroll
    for (int j = 0; j < 8; ++j) {
        v[j] = __expf(v[j] - vmax);
        sum += v[j];
    }
    red[tid] = sum;
    __syncthreads();
    for (int s = 128; s > 0; s >>= 1) {
        if (tid < s) red[tid] += red[tid + s];
        __syncthreads();
    }
    const float inv = 1.f / red[0];

    float* dst = Pi + (size_t)row * NN;
#pragma unroll
    for (int j = 0; j < 8; ++j)
        dst[tid + j * 256] = v[j] * inv;
}

// ---------------------------------------------------------------------------
// Kernel 2: PiT[n][k] = (bf16)Pi[k][n]   (LDS-tiled 64x64 transpose)
// ---------------------------------------------------------------------------
__global__ __launch_bounds__(256) void seir_transpose_bf16(const float* __restrict__ Pi,
                                                           __bf16* __restrict__ PiT) {
    __shared__ float tile[64][65];
    const int bx = blockIdx.x;           // n tile
    const int by = blockIdx.y;           // k tile
    const int tx = threadIdx.x & 63;
    const int ty = threadIdx.x >> 6;     // 0..3
#pragma unroll
    for (int j = 0; j < 16; ++j) {
        int r = ty + 4 * j;
        tile[r][tx] = Pi[(size_t)(by * 64 + r) * NN + bx * 64 + tx];
    }
    __syncthreads();
#pragma unroll
    for (int j = 0; j < 16; ++j) {
        int r = ty + 4 * j;
        // PiT[n][k], n = bx*64+r, k = by*64+tx  (coalesced store, bank-conflict-free LDS read)
        PiT[(size_t)(bx * 64 + r) * NN + by * 64 + tx] = (__bf16)tile[tx][r];
    }
}

// ---------------------------------------------------------------------------
// Grid-wide barrier (64 co-resident blocks, agent-scope atomics)
// ---------------------------------------------------------------------------
__device__ inline void grid_sync(unsigned* cnt, unsigned* gen) {
    __threadfence();          // make this thread's stores agent-visible
    __syncthreads();
    if (threadIdx.x == 0) {
        unsigned g = __hip_atomic_load(gen, __ATOMIC_RELAXED, __HIP_MEMORY_SCOPE_AGENT);
        unsigned arrived =
            __hip_atomic_fetch_add(cnt, 1u, __ATOMIC_ACQ_REL, __HIP_MEMORY_SCOPE_AGENT);
        if (arrived == NBLK - 1) {
            __hip_atomic_store(cnt, 0u, __ATOMIC_RELAXED, __HIP_MEMORY_SCOPE_AGENT);
            __hip_atomic_store(gen, g + 1u, __ATOMIC_RELEASE, __HIP_MEMORY_SCOPE_AGENT);
        } else {
            while (__hip_atomic_load(gen, __ATOMIC_ACQUIRE, __HIP_MEMORY_SCOPE_AGENT) == g)
                __builtin_amdgcn_s_sleep(2);
        }
    }
    __syncthreads();
}

// ---------------------------------------------------------------------------
// Kernel 3: persistent 271-step SEIR simulation.
// 64 blocks x 256 threads = 512 wave32 -> one 16x16 WMMA output tile each.
// The block's 32-column slice of PiT (B operand, constant across steps) is
// copied ONCE into LDS via async global->LDS b128 copies, then every step's
// matmul reads B from LDS (ds_load) and A (bf16 I state) from L2.
// ---------------------------------------------------------------------------
__global__ __launch_bounds__(256) void seir_sim(const float* __restrict__ xhist,
                                                const float* __restrict__ beta,
                                                const float* __restrict__ sigma,
                                                const float* __restrict__ gamma,
                                                float* __restrict__ out,
                                                unsigned char* __restrict__ ws) {
    __shared__ __bf16 ldsB[32][LDSB_STRIDE];     // 131,584 B of the 320 KB WGP LDS

    unsigned* sync_cnt = (unsigned*)(ws + OFF_SYNC);
    unsigned* sync_gen = sync_cnt + 1;
    float* S = (float*)(ws + OFF_S);
    float* E = (float*)(ws + OFF_E);
    float* I = (float*)(ws + OFF_I);
    __bf16* ibf0 = (__bf16*)(ws + OFF_IBF0);
    __bf16* ibf1 = (__bf16*)(ws + OFF_IBF1);
    const __bf16* PiT = (const __bf16*)(ws + OFF_PIT);

    const int wg    = blockIdx.x;            // 0..63
    const int wave  = threadIdx.x >> 5;      // 0..7
    const int lane  = threadIdx.x & 31;
    const int mi    = wave & 3;              // M tile (batch rows 16*mi..)
    const int ni    = wave >> 2;             // 0..1  (16-col half of this block)
    const int lhalf = lane >> 4;             // 0/1
    const int l15   = lane & 15;

    const int n = wg * 32 + ni * 16 + l15;   // this lane's output column
    const float bet = beta[n];
    const float sig = sigma[n];
    const float gam = gamma[n];

    // ---- one-time async fill of the block's B slice: ldsB[r][k] = PiT[32*wg+r][k]
    // 32 rows x 2048 bf16 = 8192 x 16B chunks, 32 per thread (ASYNCcnt <= 32/wave)
    {
        const __bf16* gbase = PiT + (size_t)(wg * 32) * NN;
#pragma unroll
        for (int i = 0; i < 32; ++i) {
            const int c   = threadIdx.x + 256 * i;   // 0..8191
            const int row = c >> 8;                  // 0..31
            const int j   = c & 255;                 // 16B chunk within row
            unsigned lds_addr = (unsigned)(uintptr_t)(&ldsB[row][j * 8]);
            const __bf16* g = gbase + (size_t)row * NN + j * 8;
            asm volatile("global_load_async_to_lds_b128 %0, %1, off"
                         :: "v"(lds_addr), "v"(g) : "memory");
        }
    }

    // ---- initialize state owned by this wave (overlaps with async B fill)
    __bf16* Icur = ibf0;
    __bf16* Inxt = ibf1;
#pragma unroll
    for (int r = 0; r < 8; ++r) {
        const int b = mi * 16 + r + 8 * lhalf;
        const size_t idx = (size_t)b * NN + n;
        float iv = fmaxf(xhist[((size_t)b * TT + (TT - 1)) * NN + n], 1e-6f);
        S[idx] = 0.98f;
        E[idx] = 0.0f;
        I[idx] = iv;
        Icur[idx] = (__bf16)iv;
    }

    asm volatile("s_wait_asynccnt 0x0" ::: "memory");   // B slice resident in LDS
    __syncthreads();
    grid_sync(sync_cnt, sync_gen);

    // per-lane fragment bases (A: bf16 16x32 layout from global; B: 32x16 from LDS)
    const size_t arow_off   = (size_t)(mi * 16 + l15) * NN + lhalf * 8;
    const __bf16* BrowL     = &ldsB[ni * 16 + l15][lhalf * 8];

    for (int s = 0; s < NSTEPS; ++s) {
        const __bf16* Arow = Icur + arow_off;
        v8f acc = {};
#pragma unroll 4
        for (int kk = 0; kk < NN / 32; ++kk) {
            v8bf alo = *(const v8bf*)(Arow + kk * 32);
            v8bf ahi = *(const v8bf*)(Arow + kk * 32 + 16);
            v16bf a = __builtin_shufflevector(alo, ahi, 0, 1, 2, 3, 4, 5, 6, 7,
                                              8, 9, 10, 11, 12, 13, 14, 15);
            v8bf blo = *(const v8bf*)(BrowL + kk * 32);
            v8bf bhi = *(const v8bf*)(BrowL + kk * 32 + 16);
            v16bf bm = __builtin_shufflevector(blo, bhi, 0, 1, 2, 3, 4, 5, 6, 7,
                                               8, 9, 10, 11, 12, 13, 14, 15);
            acc = __builtin_amdgcn_wmma_f32_16x16x32_bf16(
                false, a, false, bm, (short)0, acc, false, false);
        }

        // ---- elementwise SEIR update for the 8 (b,n) elements this lane owns
#pragma unroll
        for (int r = 0; r < 8; ++r) {
            const int b = mi * 16 + r + 8 * lhalf;      // C/D layout: M = r + 8*(lane>>4)
            const size_t idx = (size_t)b * NN + n;
            const float lam = acc[r] * bet;             // lam = (I @ Pi) * beta
            const float Sv = S[idx], Ev = E[idx], Iv = I[idx];
            const float E2 = Ev + bet * Sv * lam - sig * Ev;
            float I2 = fmaxf(Iv + sig * Ev - gam * Iv, 0.0f);
            const float S2 = fmaxf(1.0f - E2 - I2, 0.01f);
            S[idx] = S2;
            E[idx] = E2;
            I[idx] = I2;
            Inxt[idx] = (__bf16)I2;
            if (s >= TT - 1)                            // last `horizon` states -> output
                out[((size_t)b * HOR + (s - (TT - 1))) * NN + n] = I2;
        }

        grid_sync(sync_cnt, sync_gen);
        __bf16* t = Icur; Icur = Inxt; Inxt = t;        // swap double buffer
    }
}

// ---------------------------------------------------------------------------
extern "C" void kernel_launch(void* const* d_in, const int* in_sizes, int n_in,
                              void* d_out, int out_size, void* d_ws, size_t ws_size,
                              hipStream_t stream) {
    const float* xhist    = (const float*)d_in[0];
    const float* beta     = (const float*)d_in[1];
    const float* sigma    = (const float*)d_in[2];
    const float* gamma    = (const float*)d_in[3];
    const float* pilogits = (const float*)d_in[4];
    float* out = (float*)d_out;
    unsigned char* ws = (unsigned char*)d_ws;

    float*  Pi_out = out + OUT_I_ELEMS;          // f32 Pi lives directly in d_out
    __bf16* PiT    = (__bf16*)(ws + OFF_PIT);

    hipLaunchKernelGGL(seir_zero_sync, dim3(1), dim3(8), 0, stream,
                       (unsigned*)(ws + OFF_SYNC));
    hipLaunchKernelGGL(seir_softmax, dim3(NN), dim3(256), 0, stream,
                       pilogits, Pi_out);
    hipLaunchKernelGGL(seir_transpose_bf16, dim3(NN / 64, NN / 64), dim3(256), 0, stream,
                       Pi_out, PiT);
    hipLaunchKernelGGL(seir_sim, dim3(NBLK), dim3(256), 0, stream,
                       xhist, beta, sigma, gamma, out, ws);
}
